// LUTLayerBasic_81741817577533
// MI455X (gfx1250) — matrix operand order repaired and checked by reading, not compiled, not run on hardware
//
#include <hip/hip_runtime.h>
#include <hip/hip_bf16.h>
#include <stdint.h>

// Problem constants (match reference)
#define BB    2048   // batch
#define NIN   2048   // input features
#define DD    1024   // detectors
#define AA    4      // anchors per detector
#define OO    2048   // outputs
#define CC    16     // channels per detector (1<<A)

typedef __attribute__((ext_vector_type(16))) __bf16 v16bf;
typedef __attribute__((ext_vector_type(8)))  float  v8f;

union Frag16 {
    v16bf    v;
    uint32_t u[8];
};

// ---------------------------------------------------------------------------
// Kernel 1: per-(sample, detector) 4-bit channel codes.
// One block per sample; x row staged in LDS (8 KB), anchors read as int4.
// ---------------------------------------------------------------------------
__global__ __launch_bounds__(256)
void lut_codes_kernel(const float* __restrict__ x,
                      const int*   __restrict__ anchors,
                      unsigned char* __restrict__ codes)
{
    __shared__ float xrow[NIN];
    const int b   = blockIdx.x;
    const int tid = threadIdx.x;

    const float* xsrc = x + (size_t)b * NIN;
    #pragma unroll
    for (int i = tid; i < NIN; i += 256) xrow[i] = xsrc[i];
    __syncthreads();

    #pragma unroll
    for (int d = tid; d < DD; d += 256) {
        int4 a = ((const int4*)anchors)[d];  // 4 anchor indices, coalesced 16B
        unsigned int c = 0;
        c |= (xrow[a.x] > 0.0f) ? 1u : 0u;
        c |= (xrow[a.y] > 0.0f) ? 2u : 0u;
        c |= (xrow[a.z] > 0.0f) ? 4u : 0u;
        c |= (xrow[a.w] > 0.0f) ? 8u : 0u;
        codes[(size_t)b * DD + d] = (unsigned char)c;
    }
}

// f32 -> bf16 (round to nearest even), returned in low 16 bits
__device__ __forceinline__ uint32_t f2bf(float f)
{
    uint32_t u = __float_as_uint(f);
    return (u + 0x7FFFu + ((u >> 16) & 1u)) >> 16;
}

// ---------------------------------------------------------------------------
// Kernel 2: one-hot x weights GEMM on the bf16 WMMA pipe.
// Block = 256 threads (8 waves), tile = 128 samples x 128 outputs.
// Outer loop over groups of KSUB detector pairs (each pair = K=32 WMMA slab),
// so each barrier covers KSUB*8 = 16 WMMAs per wave.
// LDS holds 32x128 weight slabs as bf16, stored [output][k] with a padded
// stride of 20 u32 per output row (16 u32 data + 4 pad) so each lane's
// B fragment is two runs of 4 u32 (VGPR 0-3 = K 8h..8h+7, VGPR 4-7 = K 16+8h..23+8h;
// h = lane / 16), read as two ds_load_b128.
// ---------------------------------------------------------------------------
#define LDS_STRIDE 20          // u32 per output row
#define NDP  (DD / 2)          // 512 detector pairs
#define KSUB 2                 // detector pairs staged per barrier
#define NIT  (NDP / KSUB)      // 256 outer iterations

__global__ __launch_bounds__(256)
void lut_wmma_kernel(const float*         __restrict__ weights,
                     const unsigned char* __restrict__ codes,
                     float*               __restrict__ out)
{
    __shared__ uint32_t ldsW[2][KSUB][128 * LDS_STRIDE];   // 2 x 2 x 10 KB

    const int tid  = threadIdx.x;
    const int lane = tid & 31;
    const int wave = tid >> 5;          // 0..7 -> 16-sample slice
    const int hi   = lane >> 4;         // lane half (selects K range)
    const int mrow = lane & 15;         // A-matrix row / D-matrix column

    const int o0 = blockIdx.x * 128;    // output supertile
    const int b0 = blockIdx.y * 128;    // sample supertile

    // Stage weights[dp*32 .. dp*32+31][o0 .. o0+127] for KSUB pairs -> LDS.
    auto stage = [&](int buf, int it) {
        const int o_loc = tid & 127;             // 128 outputs / half-block
        const int kp0   = tid >> 7;              // 0 or 1
        #pragma unroll
        for (int sub = 0; sub < KSUB; ++sub) {
            const int dp    = it * KSUB + sub;
            const int nbase = dp * 32;
            const float* wsrc = weights + (size_t)nbase * OO + o0;
            uint32_t* dst = &ldsW[buf][sub][o_loc * LDS_STRIDE];
            #pragma unroll
            for (int kp = kp0; kp < 16; kp += 2) {   // 16 u32/row = 32 halves
                const int k = kp * 2;
                float w0 = wsrc[(size_t)k       * OO + o_loc];
                float w1 = wsrc[(size_t)(k + 1) * OO + o_loc];
                dst[kp] = f2bf(w0) | (f2bf(w1) << 16);
            }
        }
    };

    v8f acc[8];
    #pragma unroll
    for (int i = 0; i < 8; ++i) acc[i] = v8f{};

    const unsigned char* crow =
        codes + (size_t)(b0 + wave * 16 + mrow) * DD;

    stage(0, 0);

    for (int it = 0; it < NIT; ++it) {
        __syncthreads();
        if (it + 1 < NIT) stage((it + 1) & 1, it + 1);

        const int buf = it & 1;

        // 4 code bytes for this lane's A row: detectors it*4 .. it*4+3
        const uint32_t c4 = *(const uint32_t*)(crow + it * (KSUB * 2));

        #pragma unroll
        for (int sub = 0; sub < KSUB; ++sub) {
            const unsigned int c0 = (c4 >> (16 * sub))     & 0xFFu;
            const unsigned int c1 = (c4 >> (16 * sub + 8)) & 0xFFu;

            // --- one-hot A fragment (exact in bf16) ---------------------
            // lanes 0-15 : VGPR0-3 = K 0..7,  VGPR4-7 = K 16..23
            // lanes16-31 : VGPR0-3 = K 8..15, VGPR4-7 = K 24..31
            const unsigned int m0 = ((1u << c0) >> (8 * hi)) & 0xFFu;
            const unsigned int m1 = ((1u << c1) >> (8 * hi)) & 0xFFu;

            Frag16 afrag;
            #pragma unroll
            for (int i = 0; i < 4; ++i) {
                const unsigned int b2 = (m0 >> (2 * i)) & 3u;
                afrag.u[i] = ((0u - (b2 & 1u)) & 0x3F80u) |
                             ((0u - (b2 >> 1)) & 0x3F800000u);
            }
            #pragma unroll
            for (int i = 0; i < 4; ++i) {
                const unsigned int b2 = (m1 >> (2 * i)) & 3u;
                afrag.u[4 + i] = ((0u - (b2 & 1u)) & 0x3F80u) |
                                 ((0u - (b2 >> 1)) & 0x3F800000u);
            }

            // --- 8 WMMAs: 16 samples x 128 outputs, K=32 ----------------
            #pragma unroll
            for (int ot = 0; ot < 8; ++ot) {
                Frag16 bfrag;
                const uint32_t* src =
                    &ldsW[buf][sub][(ot * 16 + mrow) * LDS_STRIDE + hi * 4];
                #pragma unroll
                for (int i = 0; i < 8; ++i) bfrag.u[i] = src[i < 4 ? i : i + 4];

                acc[ot] = __builtin_amdgcn_wmma_f32_16x16x32_bf16(
                    /*neg_a=*/false, afrag.v,
                    /*neg_b=*/false, bfrag.v,
                    /*c_mod=*/(short)0, acc[ot],
                    /*reuse_a=*/false, /*reuse_b=*/false);
            }
        }
    }

    // --- store D tiles (C/D layout: VGPR r -> rows r, r+8; lane -> col) ----
    #pragma unroll
    for (int ot = 0; ot < 8; ++ot) {
        #pragma unroll
        for (int r = 0; r < 8; ++r) {
            const int row = b0 + wave * 16 + r + 8 * hi;
            out[(size_t)row * OO + o0 + ot * 16 + mrow] = acc[ot][r];
        }
    }
}

// ---------------------------------------------------------------------------
// Launch: inputs in setup_inputs() order: x [B*N_IN] f32, anchors [D*A] int,
// weights [N_LUT*O] f32. Output: [B*O] f32. Scratch: codes uint8 [B*D] (2 MB).
// ---------------------------------------------------------------------------
extern "C" void kernel_launch(void* const* d_in, const int* in_sizes, int n_in,
                              void* d_out, int out_size, void* d_ws, size_t ws_size,
                              hipStream_t stream)
{
    const float* x       = (const float*)d_in[0];
    const int*   anchors = (const int*)d_in[1];
    const float* weights = (const float*)d_in[2];
    float*       out     = (float*)d_out;
    unsigned char* codes = (unsigned char*)d_ws;

    lut_codes_kernel<<<BB, 256, 0, stream>>>(x, anchors, codes);

    dim3 grid(OO / 128, BB / 128);   // 16 x 16 supertiles
    lut_wmma_kernel<<<grid, 256, 0, stream>>>(weights, codes, out);
}
